// PGN_TRANSFORMER_12249246728309
// MI455X (gfx1250) — matrix operand order repaired and checked by reading, not compiled, not run on hardware
//
#include <hip/hip_runtime.h>
#include <math.h>

// ---------------- problem constants (match reference) ----------------
constexpr int kL = 3, kD = 512, kH = 8, kDFF = 2048, kV = 32000, kOOV = 64;
constexpr int kB = 8, kS = 512, kT = 128, kDepth = 64;
constexpr int kVExt = kV + kOOV;                  // 32064
constexpr float kEmbScale = 22.627416997969522f;  // sqrt(512)

// ---------------- WMMA vector types ----------------
typedef __attribute__((ext_vector_type(16))) __bf16       bf16x16;
typedef __attribute__((ext_vector_type(8)))  unsigned int u32x8;
typedef __attribute__((ext_vector_type(8)))  float        f32x8;

__device__ __forceinline__ unsigned short f2bf(float f) {
  unsigned int u = __builtin_bit_cast(unsigned int, f);
  unsigned int r = u + 0x7FFFu + ((u >> 16) & 1u);  // round-to-nearest-even
  return (unsigned short)(r >> 16);
}

__device__ __forceinline__ bf16x16 load_frag_a(const unsigned short* p) {
  // two contiguous 16B chunks (elems 0..7 at +0, elems 8..15 at +8 halves)
  uint4 a0 = *(const uint4*)(p);
  uint4 a1 = *(const uint4*)(p + 8);
  u32x8 w;
  w[0] = a0.x; w[1] = a0.y; w[2] = a0.z; w[3] = a0.w;
  w[4] = a1.x; w[5] = a1.y; w[6] = a1.z; w[7] = a1.w;
  return __builtin_bit_cast(bf16x16, w);
}

__device__ __forceinline__ bf16x16 load_frag_b(const unsigned short* p) {
  // 16 contiguous halves (32B)
  uint4 b0 = *(const uint4*)(p);
  uint4 b1 = *(const uint4*)(p + 8);
  u32x8 w;
  w[0] = b0.x; w[1] = b0.y; w[2] = b0.z; w[3] = b0.w;
  w[4] = b1.x; w[5] = b1.y; w[6] = b1.z; w[7] = b1.w;
  return __builtin_bit_cast(bf16x16, w);
}

__device__ __forceinline__ f32x8 wmma_bf16(bf16x16 a, bf16x16 b, f32x8 c) {
  return __builtin_amdgcn_wmma_f32_16x16x32_bf16(false, a, false, b,
                                                 (short)0, c, false, false);
}

// =====================================================================
// Pure-bf16 batched GEMM via v_wmma_f32_16x16x32_bf16, 2x2 register tiles.
//   C[M,N] = act(A[M,K] @ B^T + bias)
// A: bf16 row-major [M,K] (lda). B: bf16 row-major [N,K] (ldb) ("NT" only;
// operands pre-transposed/pre-converted outside). One wave computes a 32x32
// block (4 WMMA tiles, 2 loads per WMMA); block = 4 waves = 128 rows.
// Requires M%128==0, N%32==0, K%32==0 (all call sites comply).
// Batch offset = (batch/inner)*sXo + (batch%inner)*sXi (head slicing).
// outBf: 0 -> C is float[ldc], 1 -> C is bf16(ushort)[ldc].
// =====================================================================
__global__ __launch_bounds__(128)
void gemm_bf16_wmma_kernel(const unsigned short* __restrict__ A,
                           const unsigned short* __restrict__ Bm,
                           const float* __restrict__ bias, void* __restrict__ Cv,
                           int M, int N, int K, int lda, int ldb, int ldc,
                           long long sAo, long long sAi, long long sBo, long long sBi,
                           long long sCo, long long sCi,
                           int inner, int relu, int outBf)
{
  const int batch = blockIdx.z;
  const int bo = batch / inner, bi = batch % inner;
  A  += bo * sAo + bi * sAi;
  Bm += bo * sBo + bi * sBi;

  const int lane = threadIdx.x & 31;
  const int wave = threadIdx.x >> 5;
  const int hsel = lane >> 4;           // 0: lanes 0-15, 1: lanes 16-31
  const int l16  = lane & 15;
  const int mtile = (blockIdx.y * 4 + wave) * 32;
  const int ntile = blockIdx.x * 32;

  // A fragment layout (16-bit A 16x32, matched to B):
  //  lanes 0-15 : elems 0-15 -> K=k0+0..15
  //  lanes 16-31: elems 0-15 -> K=k0+16..31
  // B fragment (16-bit B 32x16): lanes 0-15 hold K=k0+0..15 of column ncol,
  //  lanes 16-31 hold K=k0+16..31 -> contiguous 16 halves in [N,K] storage.
  const unsigned short* ap0 = A + (size_t)(mtile + l16) * lda + hsel * 16;
  const unsigned short* ap1 = ap0 + (size_t)16 * lda;
  const unsigned short* bp0 = Bm + (size_t)(ntile + l16) * ldb + hsel * 16;
  const unsigned short* bp1 = bp0 + (size_t)16 * ldb;

  f32x8 acc00 = {}, acc01 = {}, acc10 = {}, acc11 = {};

  for (int k0 = 0; k0 < K; k0 += 32) {
    if (k0 + 32 < K) {
      __builtin_prefetch(ap0 + k0 + 64, 0, 1);
      __builtin_prefetch(ap1 + k0 + 64, 0, 1);
      __builtin_prefetch(bp0 + k0 + 64, 0, 1);
      __builtin_prefetch(bp1 + k0 + 64, 0, 1);
    }
    bf16x16 a0 = load_frag_a(ap0 + k0);
    bf16x16 a1 = load_frag_a(ap1 + k0);
    bf16x16 b0 = load_frag_b(bp0 + k0);
    bf16x16 b1 = load_frag_b(bp1 + k0);
    acc00 = wmma_bf16(a0, b0, acc00);
    acc01 = wmma_bf16(a0, b1, acc01);
    acc10 = wmma_bf16(a1, b0, acc10);
    acc11 = wmma_bf16(a1, b1, acc11);
  }

#pragma unroll
  for (int t = 0; t < 4; ++t) {
    const f32x8& a = (t == 0) ? acc00 : (t == 1) ? acc01 : (t == 2) ? acc10 : acc11;
    const int ms = (t >> 1) * 16, ns = (t & 1) * 16;
    const int col = ntile + ns + l16;
    const float bc = bias ? bias[col] : 0.0f;
    if (outBf) {
      unsigned short* C = (unsigned short*)Cv + bo * sCo + bi * sCi;
#pragma unroll
      for (int r = 0; r < 8; ++r) {
        int row = mtile + ms + hsel * 8 + r;   // C/D layout: VGPR r -> M=r / 8+r
        float v = a[r] + bc;
        if (relu) v = fmaxf(v, 0.0f);
        C[(size_t)row * ldc + col] = f2bf(v);
      }
    } else {
      float* C = (float*)Cv + bo * sCo + bi * sCi;
#pragma unroll
      for (int r = 0; r < 8; ++r) {
        int row = mtile + ms + hsel * 8 + r;
        float v = a[r] + bc;
        if (relu) v = fmaxf(v, 0.0f);
        C[(size_t)row * ldc + col] = v;
      }
    }
  }
}

// ---------------- transpose + f32->bf16: dst[C,R] = bf16(src[R,C]) ----------------
// batched via blockIdx.z (src/dst advance by R*C per batch)
__global__ void cvtT_kernel(const float* __restrict__ src, unsigned short* __restrict__ dst,
                            int R, int Cc)
{
  size_t base = (size_t)blockIdx.z * R * Cc;
  int i = blockIdx.x * blockDim.x + threadIdx.x;
  if (i >= R * Cc) return;
  int c = i / R, r = i % R;
  dst[base + i] = f2bf(src[base + (size_t)r * Cc + c]);
}

// ---------------- softmax over rows (in place), optional mask + bf16 copy ------
// maskMode 0: none; 1: mask[b*cols+k] ([B,1,1,S]); 2: mask[q*cols+k] ([T,T]).
__global__ __launch_bounds__(256)
void softmax_kernel(float* __restrict__ x, unsigned short* __restrict__ xb,
                    const float* __restrict__ mask,
                    int cols, int ldr, float scale, int maskMode, int Hh, int Tq)
{
  __shared__ float red[256];
  const int row = blockIdx.x;
  float* xr = x + (size_t)row * ldr;
  const int q = row % Tq;
  const int b = (row / Tq) / Hh;
  const int tid = threadIdx.x;

  float mx = -1e30f;
  for (int k = tid; k < cols; k += 256) {
    float mv = 0.0f;
    if (maskMode == 1)      mv = mask[(size_t)b * cols + k];
    else if (maskMode == 2) mv = mask[(size_t)q * cols + k];
    float v = xr[k] * scale + mv * -1e9f;
    xr[k] = v;
    mx = fmaxf(mx, v);
  }
  red[tid] = mx; __syncthreads();
  for (int s = 128; s > 0; s >>= 1) {
    if (tid < s) red[tid] = fmaxf(red[tid], red[tid + s]);
    __syncthreads();
  }
  mx = red[0]; __syncthreads();

  float sum = 0.0f;
  for (int k = tid; k < cols; k += 256) {
    float e = expf(xr[k] - mx);
    xr[k] = e;
    sum += e;
  }
  red[tid] = sum; __syncthreads();
  for (int s = 128; s > 0; s >>= 1) {
    if (tid < s) red[tid] += red[tid + s];
    __syncthreads();
  }
  const float inv = 1.0f / red[0];
  for (int k = tid; k < cols; k += 256) {
    float v = xr[k] * inv;
    xr[k] = v;
    if (xb) xb[(size_t)row * ldr + k] = f2bf(v);
  }
}

// ---------------- y = LayerNorm(x + r)*g + beta, dual f32 + bf16 out ----------
__global__ __launch_bounds__(256)
void add_ln_kernel(const float* __restrict__ x, const float* __restrict__ r,
                   const float* __restrict__ g, const float* __restrict__ beta,
                   float* __restrict__ y, unsigned short* __restrict__ yb, int Dd)
{
  __shared__ float s1[256], s2[256];
  const int row = blockIdx.x, tid = threadIdx.x;
  const float* xr = x + (size_t)row * Dd;
  const float* rr = r + (size_t)row * Dd;
  float a = 0.0f, b2 = 0.0f;
  for (int d = tid; d < Dd; d += 256) {
    float v = xr[d] + rr[d];
    a += v; b2 += v * v;
  }
  s1[tid] = a; s2[tid] = b2; __syncthreads();
  for (int s = 128; s > 0; s >>= 1) {
    if (tid < s) { s1[tid] += s1[tid + s]; s2[tid] += s2[tid + s]; }
    __syncthreads();
  }
  const float m   = s1[0] / Dd;
  const float var = s2[0] / Dd - m * m;
  const float inv = rsqrtf(var + 1e-6f);
  float* yr = y + (size_t)row * Dd;
  for (int d = tid; d < Dd; d += 256) {
    float v = (xr[d] + rr[d] - m) * inv * g[d] + beta[d];
    yr[d] = v;
    if (yb) yb[(size_t)row * Dd + d] = f2bf(v);
  }
}

// ---------------- embedding * sqrt(D) + positional encoding (dual out) --------
__global__ void embed_kernel(const int* __restrict__ ids, const float* __restrict__ emb,
                             float* __restrict__ xo, unsigned short* __restrict__ xb,
                             int ntok, int seqLen)
{
  int i = blockIdx.x * blockDim.x + threadIdx.x;
  if (i >= ntok * kD) return;
  int t = i / kD, d = i % kD;
  int p = t % seqLen;
  float angle = (float)p * expf(-logf(10000.0f) * (float)(2 * (d / 2)) / (float)kD);
  float pe = ((d & 1) == 0) ? sinf(angle) : cosf(angle);
  float v = emb[(size_t)ids[t] * kD + d] * kEmbScale + pe;
  xo[i] = v;
  xb[i] = f2bf(v);
}

// ---------------- p_gen = sigmoid((a+b+c)@Vw + Vb) ----------------
__global__ __launch_bounds__(128)
void pgen_kernel(const float* __restrict__ embd, const float* __restrict__ outd,
                 const float* __restrict__ ctx,
                 const float* __restrict__ Wx, const float* __restrict__ bWx,
                 const float* __restrict__ Ws, const float* __restrict__ bWs,
                 const float* __restrict__ Wh, const float* __restrict__ bWh,
                 const float* __restrict__ Vw, const float* __restrict__ Vb,
                 float* __restrict__ pg)
{
  __shared__ float red[128];
  const int row = blockIdx.x, tid = threadIdx.x;
  float s = 0.0f;
  for (int d = tid; d < kD; d += 128) {
    size_t o = (size_t)row * kD + d;
    s += embd[o] * Wx[d] + outd[o] * Ws[d] + ctx[o] * Wh[d];
  }
  red[tid] = s; __syncthreads();
  for (int st = 64; st > 0; st >>= 1) {
    if (tid < st) red[tid] += red[tid + st];
    __syncthreads();
  }
  if (tid == 0) {
    float t = (red[0] + bWx[0] + bWs[0] + bWh[0]) * Vw[0] + Vb[0];
    pg[row] = 1.0f / (1.0f + expf(-t));
  }
}

// ---------------- attn_dists[b,t,s] = mean_h block2[b,h,t,s] ----------------
__global__ void attn_dists_kernel(const float* __restrict__ cross, float* __restrict__ ad)
{
  int i = blockIdx.x * blockDim.x + threadIdx.x;
  if (i >= kB * kT * kS) return;
  int s = i % kS, t = (i / kS) % kT, b = i / (kS * kT);
  float acc = 0.0f;
  for (int h = 0; h < kH; ++h)
    acc += cross[((size_t)(b * kH + h) * kT + t) * kS + s];
  ad[i] = acc * (1.0f / kH);
}

// ---------------- logits: vocab part scaled by p_gen, OOV zeroed ----------------
__global__ void vocab_part_kernel(float* __restrict__ logits, const float* __restrict__ pg)
{
  size_t i = (size_t)blockIdx.x * blockDim.x + threadIdx.x;
  if (i >= (size_t)kB * kT * kVExt) return;
  int bt = (int)(i / kVExt), v = (int)(i % kVExt);
  logits[i] = (v < kV) ? pg[bt] * logits[i] : 0.0f;
}

// ---------------- copy distribution scatter-add ----------------
__global__ void scatter_kernel(float* __restrict__ logits, const float* __restrict__ ad,
                               const float* __restrict__ pg, const int* __restrict__ ext)
{
  int i = blockIdx.x * blockDim.x + threadIdx.x;
  if (i >= kB * kT * kS) return;
  int s = i % kS, t = (i / kS) % kT, b = i / (kS * kT);
  int bt = b * kT + t;
  int id = ext[b * kS + s];
  atomicAdd(&logits[(size_t)bt * kVExt + id], (1.0f - pg[bt]) * ad[i]);
}

// =====================================================================
// Host orchestration
// =====================================================================
static inline void gemm(hipStream_t st, const unsigned short* A, const unsigned short* Bm,
                        const float* bias, void* C,
                        int M, int N, int K, int lda, int ldb, int ldc,
                        int batches = 1, int inner = 1,
                        long long sAo = 0, long long sAi = 0,
                        long long sBo = 0, long long sBi = 0,
                        long long sCo = 0, long long sCi = 0,
                        int relu = 0, int outBf = 0)
{
  dim3 grid(N / 32, M / 128, batches);
  gemm_bf16_wmma_kernel<<<grid, 128, 0, st>>>(A, Bm, bias, C, M, N, K, lda, ldb, ldc,
                                              sAo, sAi, sBo, sBi, sCo, sCi,
                                              inner, relu, outBf);
}

static inline void cvtT(hipStream_t st, const float* src, unsigned short* dst,
                        int R, int Cc, int batches = 1)
{
  dim3 grid((R * Cc + 255) / 256, 1, batches);
  cvtT_kernel<<<grid, 256, 0, st>>>(src, dst, R, Cc);
}

extern "C" void kernel_launch(void* const* d_in, const int* in_sizes, int n_in,
                              void* d_out, int out_size, void* d_ws, size_t ws_size,
                              hipStream_t stream)
{
  (void)in_sizes; (void)n_in; (void)out_size; (void)ws_size;
  // -------- flattened input map (setup_inputs dict order, params DFS order) ----
  const int*   inp      = (const int*)d_in[0];
  const int*   ext_ids  = (const int*)d_in[1];
  const int*   tar      = (const int*)d_in[3];
  const float* enc_mask = (const float*)d_in[5];   // [B,1,1,S] zeros
  const float* la_mask  = (const float*)d_in[6];   // [1,1,T,T]
  const float* dec_mask = (const float*)d_in[7];   // [B,1,1,S] zeros
  const float* emb      = (const float*)d_in[8];   // [V,D]
  auto P = [&](int i) { return (const float*)d_in[i]; };
  // enc: 9..24  (wq bq wk bk wv bv wo bo w1 bf1 w2 bf2 g1 b1n g2 b2n)
  // dec: 25..50 (wq..bo, xwq..xbo, w1 bf1 w2 bf2 g1 b1n g2 b2n g3 b3n)
  // 51 Wx 52 bWx 53 Ws 54 bWs 55 Wh 56 bWh 57 Vw 58 Vb 59 final_w 60 final_b

  // -------- workspace bump allocator (256B aligned blocks) --------
  char* base = (char*)d_ws;
  size_t off = 0;
  auto allocF = [&](size_t n) {
    float* p = (float*)(base + off);
    off += (n * sizeof(float) + 255) & ~(size_t)255;
    return p;
  };
  auto allocH = [&](size_t n) {
    unsigned short* p = (unsigned short*)(base + off);
    off += (n * sizeof(unsigned short) + 255) & ~(size_t)255;
    return p;
  };
  const int ntokE = kB * kS, ntokD = kB * kT;
  // f32 state
  float* x_enc   = allocF((size_t)ntokE * kD);
  float* x_dec   = allocF((size_t)ntokD * kD);         // embed_dec (kept for p_gen)
  float* out_dec = allocF((size_t)ntokD * kD);
  float* vb      = allocF((size_t)ntokE * kD);
  float* attnbuf = allocF((size_t)kB * kH * kS * kS);  // enc scores (largest)
  float* crossb  = allocF((size_t)kB * kH * kT * kS);  // dec cross block2
  float* tmp     = allocF((size_t)ntokE * kD);
  float* ctxbuf  = allocF((size_t)ntokD * kD);
  float* pg      = allocF((size_t)ntokD);
  // bf16 operand mirrors
  unsigned short* x_enc_bf  = allocH((size_t)ntokE * kD);
  unsigned short* x_dec_bf  = allocH((size_t)ntokD * kD);
  unsigned short* out_bf    = allocH((size_t)ntokD * kD);
  unsigned short* qb_bf     = allocH((size_t)ntokE * kD);
  unsigned short* kb_bf     = allocH((size_t)ntokE * kD);
  unsigned short* vT_bf     = allocH((size_t)kB * kD * kS);   // [B, D, Tk]
  unsigned short* attn_bf   = allocH((size_t)kB * kH * kS * kS);
  unsigned short* cross_bf  = allocH((size_t)kB * kH * kT * kS);
  unsigned short* ao_bf     = allocH((size_t)ntokE * kD);
  unsigned short* ffn_bf    = allocH((size_t)ntokE * kDFF);
  unsigned short* encT_bf   = allocH((size_t)kB * kD * kS);   // enc_output^T
  unsigned short* wT        = allocH((size_t)kV * kD);        // weight scratch (max: final_w)

  float* logits = (float*)d_out;                       // [B*T, V+OOV]
  float* adout  = logits + (size_t)kB * kT * kVExt;    // [B*T, S]

  const long long DDs = (long long)kD * kD, Ds = kD;
  const long long W1s = (long long)kD * kDFF, W2s = (long long)kDFF * kD;

  // ================= embeddings =================
  embed_kernel<<<(ntokE * kD + 255) / 256, 256, 0, stream>>>(inp, emb, x_enc, x_enc_bf, ntokE, kS);
  embed_kernel<<<(ntokD * kD + 255) / 256, 256, 0, stream>>>(tar, emb, x_dec, x_dec_bf, ntokD, kT);

  // ================= encoder =================
  for (int i = 0; i < kL; ++i) {
    const float *wq = P(9)  + i * DDs, *bq  = P(10) + i * Ds;
    const float *wk = P(11) + i * DDs, *bk  = P(12) + i * Ds;
    const float *wv = P(13) + i * DDs, *bvp = P(14) + i * Ds;
    const float *wo = P(15) + i * DDs, *bop = P(16) + i * Ds;
    const float *w1 = P(17) + i * W1s, *bf1 = P(18) + (long long)i * kDFF;
    const float *w2 = P(19) + i * W2s, *bf2 = P(20) + i * Ds;
    const float *g1 = P(21) + i * Ds,  *b1n = P(22) + i * Ds;
    const float *g2 = P(23) + i * Ds,  *b2n = P(24) + i * Ds;

    cvtT(stream, wq, wT, kD, kD);
    gemm(stream, x_enc_bf, wT, bq,  qb_bf, ntokE, kD, kD, kD, kD, kD,
         1, 1, 0, 0, 0, 0, 0, 0, 0, /*outBf=*/1);
    cvtT(stream, wk, wT, kD, kD);
    gemm(stream, x_enc_bf, wT, bk,  kb_bf, ntokE, kD, kD, kD, kD, kD,
         1, 1, 0, 0, 0, 0, 0, 0, 0, 1);
    cvtT(stream, wv, wT, kD, kD);
    gemm(stream, x_enc_bf, wT, bvp, vb,    ntokE, kD, kD, kD, kD, kD);
    cvtT(stream, vb, vT_bf, kS, kD, kB);                 // V^T per batch: [B,D,S]

    // scores[b,h] = Q_head @ K_head^T   (M=S, N=S, K=64)
    gemm(stream, qb_bf, kb_bf, nullptr, attnbuf, kS, kS, kDepth, kD, kD, kS,
         kB * kH, kH, (long long)kS * kD, kDepth, (long long)kS * kD, kDepth,
         (long long)kH * kS * kS, (long long)kS * kS);
    softmax_kernel<<<kB * kH * kS, 256, 0, stream>>>(attnbuf, attn_bf, enc_mask,
                                                     kS, kS, 0.125f, 1, kH, kS);
    // o[b,h] = A @ V_head  via V^T  (M=S, N=64, K=S)
    gemm(stream, attn_bf, vT_bf, nullptr, ao_bf, kS, kDepth, kS, kS, kS, kD,
         kB * kH, kH, (long long)kH * kS * kS, (long long)kS * kS,
         (long long)kD * kS, (long long)kDepth * kS,
         (long long)kS * kD, kDepth, 0, 1);
    cvtT(stream, wo, wT, kD, kD);
    gemm(stream, ao_bf, wT, bop, tmp, ntokE, kD, kD, kD, kD, kD);
    add_ln_kernel<<<ntokE, 256, 0, stream>>>(x_enc, tmp, g1, b1n, x_enc, x_enc_bf, kD);

    cvtT(stream, w1, wT, kD, kDFF);
    gemm(stream, x_enc_bf, wT, bf1, ffn_bf, ntokE, kDFF, kD, kD, kD, kDFF,
         1, 1, 0, 0, 0, 0, 0, 0, /*relu=*/1, /*outBf=*/1);
    cvtT(stream, w2, wT, kDFF, kD);
    gemm(stream, ffn_bf, wT, bf2, tmp, ntokE, kD, kDFF, kDFF, kDFF, kD);
    add_ln_kernel<<<ntokE, 256, 0, stream>>>(x_enc, tmp, g2, b2n, x_enc, x_enc_bf, kD);
  }
  // enc_output^T for the final context einsum
  cvtT(stream, x_enc, encT_bf, kS, kD, kB);

  // ================= decoder =================
  hipMemcpyAsync(out_dec, x_dec, (size_t)ntokD * kD * sizeof(float),
                 hipMemcpyDeviceToDevice, stream);
  hipMemcpyAsync(out_bf, x_dec_bf, (size_t)ntokD * kD * sizeof(unsigned short),
                 hipMemcpyDeviceToDevice, stream);
  for (int i = 0; i < kL; ++i) {
    const float *wq  = P(25) + i * DDs, *bq  = P(26) + i * Ds;
    const float *wk  = P(27) + i * DDs, *bk  = P(28) + i * Ds;
    const float *wv  = P(29) + i * DDs, *bvp = P(30) + i * Ds;
    const float *wo  = P(31) + i * DDs, *bop = P(32) + i * Ds;
    const float *xwq = P(33) + i * DDs, *xbq = P(34) + i * Ds;
    const float *xwk = P(35) + i * DDs, *xbk = P(36) + i * Ds;
    const float *xwv = P(37) + i * DDs, *xbv = P(38) + i * Ds;
    const float *xwo = P(39) + i * DDs, *xbo = P(40) + i * Ds;
    const float *w1  = P(41) + i * W1s, *bf1 = P(42) + (long long)i * kDFF;
    const float *w2  = P(43) + i * W2s, *bf2 = P(44) + i * Ds;
    const float *g1  = P(45) + i * Ds,  *b1n = P(46) + i * Ds;
    const float *g2  = P(47) + i * Ds,  *b2n = P(48) + i * Ds;
    const float *g3  = P(49) + i * Ds,  *b3n = P(50) + i * Ds;

    // ---- masked self-attention (M=T) ----
    cvtT(stream, wq, wT, kD, kD);
    gemm(stream, out_bf, wT, bq,  qb_bf, ntokD, kD, kD, kD, kD, kD,
         1, 1, 0, 0, 0, 0, 0, 0, 0, 1);
    cvtT(stream, wk, wT, kD, kD);
    gemm(stream, out_bf, wT, bk,  kb_bf, ntokD, kD, kD, kD, kD, kD,
         1, 1, 0, 0, 0, 0, 0, 0, 0, 1);
    cvtT(stream, wv, wT, kD, kD);
    gemm(stream, out_bf, wT, bvp, vb,    ntokD, kD, kD, kD, kD, kD);
    cvtT(stream, vb, vT_bf, kT, kD, kB);                 // [B,D,T]

    gemm(stream, qb_bf, kb_bf, nullptr, attnbuf, kT, kT, kDepth, kD, kD, kT,
         kB * kH, kH, (long long)kT * kD, kDepth, (long long)kT * kD, kDepth,
         (long long)kH * kT * kT, (long long)kT * kT);
    softmax_kernel<<<kB * kH * kT, 256, 0, stream>>>(attnbuf, attn_bf, la_mask,
                                                     kT, kT, 0.125f, 2, kH, kT);
    gemm(stream, attn_bf, vT_bf, nullptr, ao_bf, kT, kDepth, kT, kT, kT, kD,
         kB * kH, kH, (long long)kH * kT * kT, (long long)kT * kT,
         (long long)kD * kT, (long long)kDepth * kT,
         (long long)kT * kD, kDepth, 0, 1);
    cvtT(stream, wo, wT, kD, kD);
    gemm(stream, ao_bf, wT, bop, tmp, ntokD, kD, kD, kD, kD, kD);
    add_ln_kernel<<<ntokD, 256, 0, stream>>>(out_dec, tmp, g1, b1n, out_dec, out_bf, kD);

    // ---- cross-attention (K/V from encoder output, M=T, keys=S) ----
    cvtT(stream, xwq, wT, kD, kD);
    gemm(stream, out_bf, wT, xbq, qb_bf, ntokD, kD, kD, kD, kD, kD,
         1, 1, 0, 0, 0, 0, 0, 0, 0, 1);
    cvtT(stream, xwk, wT, kD, kD);
    gemm(stream, x_enc_bf, wT, xbk, kb_bf, ntokE, kD, kD, kD, kD, kD,
         1, 1, 0, 0, 0, 0, 0, 0, 0, 1);
    cvtT(stream, xwv, wT, kD, kD);
    gemm(stream, x_enc_bf, wT, xbv, vb, ntokE, kD, kD, kD, kD, kD);
    cvtT(stream, vb, vT_bf, kS, kD, kB);                 // [B,D,S]

    gemm(stream, qb_bf, kb_bf, nullptr, crossb, kT, kS, kDepth, kD, kD, kS,
         kB * kH, kH, (long long)kT * kD, kDepth, (long long)kS * kD, kDepth,
         (long long)kH * kT * kS, (long long)kT * kS);
    softmax_kernel<<<kB * kH * kT, 256, 0, stream>>>(crossb, cross_bf, dec_mask,
                                                     kS, kS, 0.125f, 1, kH, kT);
    gemm(stream, cross_bf, vT_bf, nullptr, ao_bf, kT, kDepth, kS, kS, kS, kD,
         kB * kH, kH, (long long)kH * kT * kS, (long long)kT * kS,
         (long long)kD * kS, (long long)kDepth * kS,
         (long long)kT * kD, kDepth, 0, 1);
    cvtT(stream, xwo, wT, kD, kD);
    gemm(stream, ao_bf, wT, xbo, tmp, ntokD, kD, kD, kD, kD, kD);
    add_ln_kernel<<<ntokD, 256, 0, stream>>>(out_dec, tmp, g2, b2n, out_dec, out_bf, kD);

    // ---- FFN ----
    cvtT(stream, w1, wT, kD, kDFF);
    gemm(stream, out_bf, wT, bf1, ffn_bf, ntokD, kDFF, kD, kD, kD, kDFF,
         1, 1, 0, 0, 0, 0, 0, 0, 1, 1);
    cvtT(stream, w2, wT, kDFF, kD);
    gemm(stream, ffn_bf, wT, bf2, tmp, ntokD, kD, kDFF, kDFF, kDFF, kD);
    add_ln_kernel<<<ntokD, 256, 0, stream>>>(out_dec, tmp, g3, b3n, out_dec, out_bf, kD);
  }

  // ============ pointer-generator head ============
  // context[b,t,h,:] = block2[b,h,t,:] @ enc_head[b,h]   (M=T, N=64, K=S)
  gemm(stream, cross_bf, encT_bf, nullptr, ctxbuf, kT, kDepth, kS, kS, kS, kD,
       kB * kH, kH, (long long)kH * kT * kS, (long long)kT * kS,
       (long long)kD * kS, (long long)kDepth * kS,
       (long long)kT * kD, kDepth);
  pgen_kernel<<<ntokD, 128, 0, stream>>>(x_dec, out_dec, ctxbuf,
                                         P(51), P(52), P(53), P(54),
                                         P(55), P(56), P(57), P(58), pg);

  // vocab logits straight into d_out rows of width V+OOV, softmax in place
  cvtT(stream, P(59), wT, kD, kV);                       // final_w^T: [V,D]
  gemm(stream, out_bf, wT, P(60), logits, ntokD, kV, kD, kD, kD, kVExt);
  softmax_kernel<<<ntokD, 256, 0, stream>>>(logits, nullptr, nullptr,
                                            kV, kVExt, 1.0f, 0, 1, 1);

  attn_dists_kernel<<<(kB * kT * kS + 255) / 256, 256, 0, stream>>>(crossb, adout);
  vocab_part_kernel<<<(int)(((size_t)kB * kT * kVExt + 255) / 256), 256, 0, stream>>>(logits, pg);
  scatter_kernel<<<(kB * kT * kS + 255) / 256, 256, 0, stream>>>(logits, adout, pg, ext_ids);
}
